// GraphAttention_60344290509408
// MI455X (gfx1250) — matrix operand order repaired and checked by reading, not compiled, not run on hardware
//
#include <hip/hip_runtime.h>
#include <hip/hip_bf16.h>

typedef __attribute__((ext_vector_type(16))) _Float16 v16h;
typedef __attribute__((ext_vector_type(8)))  _Float16 v8h;
typedef __attribute__((ext_vector_type(8)))  float    v8f;

#define TT 80
#define AA 80
#define NN (TT*AA)
#define DD 128
#define HH 4
#define CC 32
#define BM 64    // GEMM M-rows per block

// padded LDS row strides (in halves): 16B-aligned, non-degenerate bank stride
#define XS 136   // 272B rows for gemm operands (68-bank stride, conflict-free frags)
#define AS 104   // 208B rows for attend operands (52-bank stride, conflict-free frags)

__device__ __forceinline__ float lrelu(float x) { return x > 0.f ? x : 0.2f * x; }

// A-matrix 16x32 f16 fragment (ISA 7.12.2): lane L holds row M=L%16;
// lanes 0-15 K-runs {0..7, 16..23}, lanes 16-31 K-runs {8..15, 24..31}.
__device__ __forceinline__ v16h load_a_frag(const _Float16* row, int kt, int lane) {
    int k0 = kt * 32 + ((lane & 16) ? 8 : 0);
    v16h a;
    ((v8h*)&a)[0] = *(const v8h*)(row + k0);        // K = k0 .. k0+7
    ((v8h*)&a)[1] = *(const v8h*)(row + k0 + 16);   // K = k0+16 .. k0+23
    return a;
}
// B-matrix 32x16 f16 fragment: lane L holds col N=L%16; lanes 0-15 K-runs {0..7, 16..23},
// lanes 16-31 K-runs {8..15, 24..31}.  row = this lane's N column stored K-major.
__device__ __forceinline__ v16h load_b_frag(const _Float16* row, int kt, int lane) {
    int k0 = kt * 32 + ((lane & 16) ? 8 : 0);
    v16h b;
    ((v8h*)&b)[0] = *(const v8h*)(row + k0);        // K = k0 .. k0+7
    ((v8h*)&b)[1] = *(const v8h*)(row + k0 + 16);   // K = k0+16 .. k0+23
    return b;
}

// ---------------------------------------------------------------------------
// Prep: W[k][n] (f32) -> WT[n][k] (f16), done once per layer.
// ---------------------------------------------------------------------------
__global__ void convert_wt(const float* __restrict__ W, _Float16* __restrict__ WT)
{
    int idx = blockIdx.x * blockDim.x + threadIdx.x;
    if (idx >= DD * DD) return;
    int k = idx / DD, n = idx % DD;       // coalesced read
    WT[n * DD + k] = (_Float16)W[idx];
}

// Prep: f32 -> f16 elementwise (layer-1 input only).
__global__ void convert_f16(const float* __restrict__ src, _Float16* __restrict__ dst,
                            int count)
{
    int idx = blockIdx.x * blockDim.x + threadIdx.x;
    if (idx < count) dst[idx] = (_Float16)src[idx];
}

// ---------------------------------------------------------------------------
// K1: xw = X @ W  (M=6400, N=128, K=128), f16 WMMA, f32 accumulate.
// 64-row M block; 8 waves x 4 M-tiles x 4 K-steps. All-f16 staging via b128 copies.
// ---------------------------------------------------------------------------
__global__ __launch_bounds__(256) void gemm_xw(const _Float16* __restrict__ Xg,
                                               const _Float16* __restrict__ WTg,
                                               float* __restrict__ xw,
                                               _Float16* __restrict__ xwh)
{
    __shared__ _Float16 WhT[DD][XS];   // WhT[n][k] = W[k][n]
    __shared__ _Float16 Xh[BM][XS];
    const int tid = threadIdx.x;
    const int nb  = blockIdx.x * BM;

    for (int idx = tid; idx < DD * (DD / 8); idx += 256) {      // 2048 v8h chunks
        int n = idx / (DD / 8), ch = idx % (DD / 8);
        *(v8h*)&WhT[n][ch * 8] = *(const v8h*)&WTg[n * DD + ch * 8];
    }
    for (int idx = tid; idx < BM * (DD / 8); idx += 256) {      // 1024 v8h chunks
        int m = idx / (DD / 8), ch = idx % (DD / 8);
        *(v8h*)&Xh[m][ch * 8] = *(const v8h*)&Xg[(size_t)(nb + m) * DD + ch * 8];
    }
    __syncthreads();

    const int lane = tid & 31;
    const int nt   = tid >> 5;                   // 0..7 N-tile
    const _Float16* brow = &WhT[nt * 16 + (lane & 15)][0];
    const int colf  = nt * 16 + (lane & 15);
    const int mbase = (lane & 16) ? 8 : 0;

#pragma unroll
    for (int mt = 0; mt < BM / 16; ++mt) {
        const _Float16* arow = &Xh[mt * 16 + (lane & 15)][0];
        v8f acc = {};
#pragma unroll
        for (int kt = 0; kt < 4; ++kt) {
            v16h a = load_a_frag(arow, kt, lane);
            v16h b = load_b_frag(brow, kt, lane);
            acc = __builtin_amdgcn_wmma_f32_16x16x32_f16(false, a, false, b,
                                                         (short)0, acc, false, false);
        }
#pragma unroll
        for (int v = 0; v < 8; ++v) {
            int node = nb + mt * 16 + mbase + v;
            float val = acc[v];
            xw [(size_t)node * DD + colf] = val;
            xwh[(size_t)node * DD + colf] = (_Float16)val;
        }
    }
}

// ---------------------------------------------------------------------------
// K2: per-(node, head) attention logits as = <xw, a_src>, ad = <xw, a_dst>
// ---------------------------------------------------------------------------
__global__ void node_alphas(const float* __restrict__ xw,
                            const float* __restrict__ a_src,
                            const float* __restrict__ a_dst,
                            float* __restrict__ asb, float* __restrict__ adb)
{
    int t = blockIdx.x * blockDim.x + threadIdx.x;
    if (t >= NN * HH) return;
    int n = t >> 2, h = t & 3;
    const float* xp = xw + (size_t)n * DD + h * CC;
    float s = 0.f, d = 0.f;
    for (int c = 0; c < CC; ++c) {
        float v = xp[c];
        s += v * a_src[h * CC + c];
        d += v * a_dst[h * CC + c];
    }
    asb[t] = s;
    adb[t] = d;
}

// ---------------------------------------------------------------------------
// K3: per-destination segment-softmax stats over row ∪ column (self once).
// ---------------------------------------------------------------------------
__global__ void softmax_stats(const float* __restrict__ asb,
                              const float* __restrict__ adb,
                              float* __restrict__ mb, float* __restrict__ den)
{
    int t = blockIdx.x * blockDim.x + threadIdx.x;
    if (t >= NN * HH) return;
    int d = t >> 2, h = t & 3;
    int r = d / AA, c = d % AA;
    float adv = adb[t];

    float mx = -3.4e38f;
    for (int c2 = 0; c2 < AA; ++c2)
        mx = fmaxf(mx, lrelu(asb[(r * AA + c2) * HH + h] + adv));
    for (int r2 = 0; r2 < TT; ++r2)
        mx = fmaxf(mx, lrelu(asb[(r2 * AA + c) * HH + h] + adv));

    float s = 0.f;
    for (int c2 = 0; c2 < AA; ++c2)
        s += __expf(lrelu(asb[(r * AA + c2) * HH + h] + adv) - mx);
    for (int r2 = 0; r2 < TT; ++r2)
        s += __expf(lrelu(asb[(r2 * AA + c) * HH + h] + adv) - mx);
    s -= __expf(lrelu(asb[t] + adv) - mx);   // self was counted in both scans

    mb[t]  = mx;
    den[t] = s;
}

// ---------------------------------------------------------------------------
// K4: structured aggregation as batched WMMA attention matmuls.
// PASS 0: per (row r, head h):  (80x80 weights) @ (80x32 features) -> write
// PASS 1: per (col c, head h):  same, self edge excluded, accumulate (+=)
// ---------------------------------------------------------------------------
template <int PASS>
__global__ __launch_bounds__(320) void attend(const _Float16* __restrict__ xwh,
                                              const float* __restrict__ asb,
                                              const float* __restrict__ adb,
                                              const float* __restrict__ mb,
                                              float* __restrict__ agg)
{
    __shared__ _Float16 At[80][AS];    // attention weights, K zero-padded to 96
    __shared__ _Float16 BtT[32][AS];   // BtT[cc][K] = xw feature cc of source K
    const int tid   = threadIdx.x;
    const int outer = blockIdx.x >> 2;   // r (PASS 0) or c (PASS 1)
    const int h     = blockIdx.x & 3;

    for (int idx = tid; idx < 80 * 96; idx += 320) {
        int M = idx / 96, K = idx % 96;
        float w = 0.f;
        if (K < 80 && !(PASS == 1 && K == M)) {   // col pass skips self (row pass has it)
            int dn = (PASS == 0) ? (outer * AA + M) : (M * AA + outer);
            int sn = (PASS == 0) ? (outer * AA + K) : (K * AA + outer);
            w = __expf(lrelu(asb[sn * HH + h] + adb[dn * HH + h]) - mb[dn * HH + h]);
        }
        At[M][K] = (_Float16)w;
    }
    for (int idx = tid; idx < 96 * 32; idx += 320) {
        int K = idx / 32, cc = idx % 32;          // coalesced global read
        _Float16 v = (_Float16)0.f;
        if (K < 80) {
            int sn = (PASS == 0) ? (outer * AA + K) : (K * AA + outer);
            v = xwh[(size_t)sn * DD + h * CC + cc];
        }
        BtT[cc][K] = v;
    }
    __syncthreads();

    const int lane = tid & 31;
    const int wv   = tid >> 5;    // 0..9
    const int mt   = wv >> 1;     // 0..4
    const int ntl  = wv & 1;      // 0..1
    const _Float16* arow = &At[mt * 16 + (lane & 15)][0];
    const _Float16* brow = &BtT[ntl * 16 + (lane & 15)][0];
    v8f acc = {};
#pragma unroll
    for (int kt = 0; kt < 3; ++kt) {
        v16h a = load_a_frag(arow, kt, lane);
        v16h b = load_b_frag(brow, kt, lane);
        acc = __builtin_amdgcn_wmma_f32_16x16x32_f16(false, a, false, b,
                                                     (short)0, acc, false, false);
    }
    const int cc    = ntl * 16 + (lane & 15);
    const int mbase = mt * 16 + ((lane & 16) ? 8 : 0);
#pragma unroll
    for (int v = 0; v < 8; ++v) {
        int M  = mbase + v;
        int dn = (PASS == 0) ? (outer * AA + M) : (M * AA + outer);
        size_t idx = (size_t)dn * DD + h * CC + cc;
        if (PASS == 0) agg[idx] = acc[v];
        else           agg[idx] += acc[v];
    }
}

// ---------------------------------------------------------------------------
// K5: normalize, add bias, ELU; emit f32 result + f16 copy (next layer input).
// ---------------------------------------------------------------------------
__global__ void combine(const float* __restrict__ agg,
                        const float* __restrict__ den,
                        const float* __restrict__ bias,
                        float* __restrict__ out, _Float16* __restrict__ outh)
{
    int idx = blockIdx.x * blockDim.x + threadIdx.x;
    if (idx >= NN * DD) return;
    int n = idx / DD, f = idx % DD, h = f / CC;
    float val = agg[idx] / den[n * HH + h] + bias[f];
    val = val > 0.f ? val : __expf(val) - 1.f;
    out[idx]  = val;
    outh[idx] = (_Float16)val;
}

// ---------------------------------------------------------------------------
extern "C" void kernel_launch(void* const* d_in, const int* in_sizes, int n_in,
                              void* d_out, int out_size, void* d_ws, size_t ws_size,
                              hipStream_t stream)
{
    const float* x    = (const float*)d_in[0];
    const float* W1   = (const float*)d_in[1];
    const float* asr1 = (const float*)d_in[2];
    const float* ads1 = (const float*)d_in[3];
    const float* b1   = (const float*)d_in[4];
    const float* W2   = (const float*)d_in[5];
    const float* asr2 = (const float*)d_in[6];
    const float* ads2 = (const float*)d_in[7];
    const float* b2   = (const float*)d_in[8];
    float* out = (float*)d_out;

    // workspace layout (~13.5 MB total, all L2-resident)
    char* w = (char*)d_ws;
    float*    xw   = (float*)   (w + 0);         // 6400*128 f32
    _Float16* xwh  = (_Float16*)(w + 3276800);   // 6400*128 f16
    float*    asb  = (float*)   (w + 4915200);   // 6400*4 f32
    float*    adb  = (float*)   (w + 5017600);
    float*    mb   = (float*)   (w + 5120000);
    float*    den  = (float*)   (w + 5222400);
    float*    agg  = (float*)   (w + 5324800);   // 6400*128 f32
    float*    hb   = (float*)   (w + 8601600);   // 6400*128 f32 (layer-1 f32 out)
    _Float16* xh   = (_Float16*)(w + 11878400);  // 6400*128 f16 (gemm input)
    _Float16* whtg = (_Float16*)(w + 13516800);  // 128*128 f16 (W^T, per layer)

    convert_f16<<<(NN * DD + 255) / 256, 256, 0, stream>>>(x, xh, NN * DD);

    for (int layer = 0; layer < 2; ++layer) {
        const float* Wm  = layer ? W2   : W1;
        const float* av  = layer ? asr2 : asr1;
        const float* dv  = layer ? ads2 : ads1;
        const float* bv  = layer ? b2   : b1;
        float* dst = layer ? out : hb;

        convert_wt   <<<(DD * DD + 255) / 256, 256, 0, stream>>>(Wm, whtg);
        gemm_xw      <<<NN / BM, 256, 0, stream>>>(xh, whtg, xw, xwh);
        node_alphas  <<<(NN * HH + 255) / 256, 256, 0, stream>>>(xw, av, dv, asb, adb);
        softmax_stats<<<(NN * HH + 255) / 256, 256, 0, stream>>>(asb, adb, mb, den);
        attend<0>    <<<TT * HH, 320, 0, stream>>>(xwh, asb, adb, mb, agg);
        attend<1>    <<<AA * HH, 320, 0, stream>>>(xwh, asb, adb, mb, agg);
        // writes layer output f32 (hb/out) and f16 (xh = next layer's gemm input)
        combine      <<<(NN * DD + 255) / 256, 256, 0, stream>>>(agg, den, bv, dst, xh);
    }
}
